// CRF_3856880632062
// MI455X (gfx1250) — compile-verified, matrix operand order checked by reading, not run on hardware
//
#include <hip/hip_runtime.h>
#include <hip/hip_bf16.h>
#include <stdint.h>

typedef __attribute__((ext_vector_type(16))) _Float16 v16h;
typedef __attribute__((ext_vector_type(8)))  float    v8f;

#define B_ 512
#define S_ 512
#define T_ 128

#define LOG2E_F 1.44269504088896340736f
#define LN2_F   0.69314718055994530942f

// fast e^x / ln(x): raw v_exp_f32 / v_log_f32 (base-2) + one multiply.
__device__ __forceinline__ float fast_exp(float x) {
    return __builtin_amdgcn_exp2f(x * LOG2E_F);
}
__device__ __forceinline__ float fast_log(float x) {
    return __builtin_amdgcn_logf(x) * LN2_F;
}

// ---- DPP16 cross-lane reduction over 16-lane rows (no LDS, no counters) ----
template <int CTRL>
__device__ __forceinline__ float dpp_movf(float x) {
    return __builtin_bit_cast(float,
        __builtin_amdgcn_update_dpp(0, __builtin_bit_cast(int, x),
                                    CTRL, 0xf, 0xf, true));
}
__device__ __forceinline__ float row16_max(float x) {
    x = fmaxf(x, dpp_movf<0xB1>(x));    // quad_perm [1,0,3,2] : xor 1
    x = fmaxf(x, dpp_movf<0x4E>(x));    // quad_perm [2,3,0,1] : xor 2
    x = fmaxf(x, dpp_movf<0x141>(x));   // row_half_mirror    : xor 7 in 8
    x = fmaxf(x, dpp_movf<0x140>(x));   // row_mirror         : xor 15 in 16
    return x;
}
__device__ __forceinline__ float row16_sum(float x) {
    x += dpp_movf<0xB1>(x);
    x += dpp_movf<0x4E>(x);
    x += dpp_movf<0x141>(x);
    x += dpp_movf<0x140>(x);
    return x;
}

// ---------------------------------------------------------------------------
// Forward algorithm in exp-domain. One workgroup = 16 batch rows, 8 waves.
// Each wave owns one 16-col tile: alpha_next = log( expA(16x128)@expT(128x16) )
// expT fragments stay in VGPRs for the whole scan. Emissions for step t+1 are
// preloaded into registers during step t; prefetch runs 4 steps ahead.
// 2 barriers per step; rowmax via DPP16 butterflies; dual WMMA accumulators.
// ---------------------------------------------------------------------------
__global__ __launch_bounds__(256) void crf_forward_kernel(
    const float* __restrict__ emissions,
    const float* __restrict__ start_transitions,
    const float* __restrict__ end_transitions,
    const float* __restrict__ transitions,
    const int*   __restrict__ mask,
    float*       __restrict__ denom)
{
    __shared__ float alpha[16][T_];                         // 8 KB f32 alpha tile
    __shared__ __align__(16) unsigned int afrag[4][32][12]; // f16 A frags, 48B/lane
    __shared__ float rowmaxLDS[16];
    __shared__ unsigned int maskbits[16][S_ / 32];          // 1 KB packed mask

    const int tid  = threadIdx.x;
    const int lane = tid & 31;
    const int wave = tid >> 5;
    const int r    = tid >> 4;      // row 0..15   (256-thread phases)
    const int c    = tid & 15;      // chunk 0..15 (8 cols each)
    const int b0   = blockIdx.x * 16;

    // ---- B fragments: exp(transitions), K-major per lane, VGPR-resident ----
    const int nt    = wave;                 // n-tile 0..7
    const int n     = lane & 15;
    const int ncol  = nt * 16 + n;
    const int koff  = (lane >= 16) ? 8 : 0;
    const int mbase = (lane >= 16) ? 8 : 0;
    v16h bfrag[4];
    #pragma unroll
    for (int f = 0; f < 4; ++f) {
        #pragma unroll
        for (int e = 0; e < 16; ++e) {
            int K = f * 32 + ((e >> 3) << 4) + koff + (e & 7);
            bfrag[f][e] = (_Float16)fast_exp(transitions[K * T_ + ncol]);
        }
    }

    // ---- pack mask bits for this block's 16 rows: 16 x (512/32) words ----
    {
        unsigned int w = 0;
        const int* mrow = mask + (size_t)(b0 + r) * S_ + c * 32;
        #pragma unroll
        for (int i = 0; i < 32; ++i) w |= (mrow[i] ? 1u : 0u) << i;
        maskbits[r][c] = w;
    }

    // ---- alpha0 = start + emissions[:,0] ----
    #pragma unroll
    for (int k = 0; k < 8; ++k) {
        int col = c * 8 + k;
        alpha[r][col] = start_transitions[col] +
                        emissions[(size_t)(b0 + r) * S_ * T_ + col];
    }

    // ---- preload emissions for t = 1 into registers ----
    float em_reg[8];
    #pragma unroll
    for (int v = 0; v < 8; ++v) {
        int m = v + mbase;
        em_reg[v] = emissions[((size_t)(b0 + m) * S_ + 1) * T_ + ncol];
    }
    __syncthreads();

    // ---- scan t = 1 .. S-1 ----
    for (int t = 1; t < S_; ++t) {
        // Phase A: rowmax (DPP butterfly), exp, pack f16 A fragments.
        // oldv/mask gathers issued early (alpha[m][ncol] is only ever
        // written by this same thread in phase B -> race-free).
        float    oldv[8];
        unsigned mbit[8];
        #pragma unroll
        for (int v = 0; v < 8; ++v) {
            int m   = v + mbase;
            oldv[v] = alpha[m][ncol];
            mbit[v] = (maskbits[m][t >> 5] >> (t & 31)) & 1u;
        }
        float a8[8];
        float mx = -3.4e38f;
        #pragma unroll
        for (int k = 0; k < 8; ++k) {
            a8[k] = alpha[r][c * 8 + k];
            mx = fmaxf(mx, a8[k]);
        }
        mx = row16_max(mx);                  // all 16 lanes of row hold rowmax
        if ((lane & 15) == 0) rowmaxLDS[r] = mx;
        {
            union { _Float16 h[8]; uint4 u; } pk;
            #pragma unroll
            for (int k = 0; k < 8; ++k)
                pk.h[k] = (_Float16)fast_exp(a8[k] - mx);
            int f   = c >> 2;
            int ln  = r + ((c & 1) << 4);
            int dwb = ((c >> 1) & 1) * 4;
            *(uint4*)&afrag[f][ln][dwb] = pk.u;   // one ds_store_b128
        }
        __syncthreads();

        // Phase B: two independent WMMA chains, merge, masked writeback.
        float rmx[8];
        #pragma unroll
        for (int v = 0; v < 8; ++v) rmx[v] = rowmaxLDS[v + mbase];

        union { uint4 q[2]; v16h v; } a0, a1, a2, a3;
        a0.q[0] = *(const uint4*)&afrag[0][lane][0];
        a0.q[1] = *(const uint4*)&afrag[0][lane][4];
        a1.q[0] = *(const uint4*)&afrag[1][lane][0];
        a1.q[1] = *(const uint4*)&afrag[1][lane][4];
        a2.q[0] = *(const uint4*)&afrag[2][lane][0];
        a2.q[1] = *(const uint4*)&afrag[2][lane][4];
        a3.q[0] = *(const uint4*)&afrag[3][lane][0];
        a3.q[1] = *(const uint4*)&afrag[3][lane][4];

        v8f acc0 = {}, acc1 = {};
        acc0 = __builtin_amdgcn_wmma_f32_16x16x32_f16(
                 false, a0.v, false, bfrag[0], (short)0, acc0, false, false);
        acc1 = __builtin_amdgcn_wmma_f32_16x16x32_f16(
                 false, a1.v, false, bfrag[1], (short)0, acc1, false, false);
        acc0 = __builtin_amdgcn_wmma_f32_16x16x32_f16(
                 false, a2.v, false, bfrag[2], (short)0, acc0, false, false);
        acc1 = __builtin_amdgcn_wmma_f32_16x16x32_f16(
                 false, a3.v, false, bfrag[3], (short)0, acc1, false, false);

        #pragma unroll
        for (int v = 0; v < 8; ++v) {
            int m = v + mbase;
            float nv = fast_log(acc0[v] + acc1[v]) + rmx[v] + em_reg[v];
            alpha[m][ncol] = mbit[v] ? nv : oldv[v];
        }

        // Phase C: pipeline emissions for t+1; prefetch t+4.
        if (t + 1 < S_) {
            #pragma unroll
            for (int v = 0; v < 8; ++v) {
                int m = v + mbase;
                const float* ep = emissions +
                    ((size_t)(b0 + m) * S_ + (t + 1)) * T_ + ncol;
                em_reg[v] = *ep;
                __builtin_prefetch(ep + 3 * T_, 0, 3);
            }
        }
        __syncthreads();
    }

    // ---- denominator[b] = logsumexp_j(alpha[b,j] + end[j]) : DPP-only ----
    float av[8];
    float mxf = -3.4e38f;
    #pragma unroll
    for (int k = 0; k < 8; ++k) {
        av[k] = alpha[r][c * 8 + k] + end_transitions[c * 8 + k];
        mxf = fmaxf(mxf, av[k]);
    }
    mxf = row16_max(mxf);
    float ss = 0.f;
    #pragma unroll
    for (int k = 0; k < 8; ++k) ss += fast_exp(av[k] - mxf);
    ss = row16_sum(ss);
    if ((lane & 15) == 0) denom[b0 + r] = mxf + fast_log(ss);
}

// ---------------------------------------------------------------------------
// Numerator: one block per batch element, block-parallel over S.
// ---------------------------------------------------------------------------
__global__ __launch_bounds__(256) void crf_numer_kernel(
    const float* __restrict__ emissions,
    const float* __restrict__ startt,
    const float* __restrict__ endt,
    const float* __restrict__ transitions,
    const int*   __restrict__ tags,
    const int*   __restrict__ mask,
    float*       __restrict__ numer)
{
    const int b   = blockIdx.x;
    const int tid = threadIdx.x;
    const int*   tg = tags + (size_t)b * S_;
    const int*   mk = mask + (size_t)b * S_;
    const float* em = emissions + (size_t)b * S_ * T_;

    float s   = 0.f;
    int   cnt = 0;
    for (int i = tid; i < S_; i += 256) {
        int m   = mk[i];
        int cur = tg[i];
        cnt += m;
        if (i == 0) {
            s += startt[cur] + em[cur];
        } else {
            float v = transitions[tg[i - 1] * T_ + cur] + em[(size_t)i * T_ + cur];
            s += m ? v : 0.f;
        }
    }
    __shared__ float sb[256];
    __shared__ int   cb[256];
    sb[tid] = s; cb[tid] = cnt;
    __syncthreads();
    for (int off = 128; off > 0; off >>= 1) {
        if (tid < off) { sb[tid] += sb[tid + off]; cb[tid] += cb[tid + off]; }
        __syncthreads();
    }
    if (tid == 0) numer[b] = sb[0] + endt[tg[cb[0] - 1]];
}

// ---------------------------------------------------------------------------
// mean(denom - numer)
// ---------------------------------------------------------------------------
__global__ __launch_bounds__(256) void crf_reduce_kernel(
    const float* __restrict__ denom, const float* __restrict__ numer,
    float* __restrict__ out)
{
    __shared__ float buf[256];
    float s = 0.f;
    for (int i = threadIdx.x; i < B_; i += 256) s += denom[i] - numer[i];
    buf[threadIdx.x] = s;
    __syncthreads();
    for (int off = 128; off > 0; off >>= 1) {
        if (threadIdx.x < off) buf[threadIdx.x] += buf[threadIdx.x + off];
        __syncthreads();
    }
    if (threadIdx.x == 0) out[0] = buf[0] / (float)B_;
}

extern "C" void kernel_launch(void* const* d_in, const int* in_sizes, int n_in,
                              void* d_out, int out_size, void* d_ws, size_t ws_size,
                              hipStream_t stream) {
    const float* emissions = (const float*)d_in[0];
    const float* startt    = (const float*)d_in[1];
    const float* endt      = (const float*)d_in[2];
    const float* trans     = (const float*)d_in[3];
    const int*   tags      = (const int*)  d_in[4];
    const int*   mask      = (const int*)  d_in[5];
    float*       out       = (float*)d_out;

    float* numer = (float*)d_ws;       // B_ floats
    float* denom = numer + B_;         // B_ floats

    crf_numer_kernel  <<<dim3(B_),      dim3(256), 0, stream>>>(
        emissions, startt, endt, trans, tags, mask, numer);
    crf_forward_kernel<<<dim3(B_ / 16), dim3(256), 0, stream>>>(
        emissions, startt, endt, trans, mask, denom);
    crf_reduce_kernel <<<dim3(1),       dim3(256), 0, stream>>>(
        denom, numer, out);
}
